// LSTMModel_76862734729309
// MI455X (gfx1250) — matrix operand order repaired and checked by reading, not compiled, not run on hardware
//
#include <hip/hip_runtime.h>
#include <hip/hip_bf16.h>

// ============================================================================
// Fused 2-layer LSTM (B=2048, T=1000, I=13, H=32) + MLP head for MI455X.
//
// Design (latency-bound recurrence, not FLOP-bound):
//  * one wave32 = one 16-row batch tile x one 16-col gate half -> 256 chains
//  * recurrent + input matmuls via V_WMMA_F32_16X16X32_F16 (K=32 == H exact)
//  * both layers fused per timestep: hs0 never touches HBM (saves ~0.5 GB)
//  * all 16 WMMA B-operands pinned in VGPRs (128 VGPRs); with ~1 wave/SIMD
//    device-wide, register pressure is free -- but we stay under 256 live
//    VGPRs to avoid s_set_vgpr_msb churn:
//      - WMMA C input = inline 0 (no bias broadcast vectors)
//      - bias folded into the activation FMA in exp2-domain:
//        sigmoid(x+b) = rcp(1 + exp2(-log2e*x + (-log2e*b)))
//        tanh(x+b)    = 1 - 2*rcp(1 + exp2(2log2e*x + (2log2e*b)))
//  * c-state fp32 in VGPRs; h transposed D-layout -> A-layout through LDS
// ============================================================================

typedef __attribute__((ext_vector_type(16))) _Float16 v16h;
typedef __attribute__((ext_vector_type(8)))  _Float16 v8h;
typedef __attribute__((ext_vector_type(8)))  float    v8f;

#define B_SZ 2048
#define T_SZ 1000
#define I_SZ 13
#define H_SZ 32
#define C_SZ 3

#define LOG2E  1.4426950408889634f
#define LOG2E2 2.8853900817779268f

static __device__ __forceinline__ v8f wmma_f16(v16h a, v16h b, v8f c) {
  // D = A(16x32 f16) * B(32x16 f16) + C(16x16 f32)
  return __builtin_amdgcn_wmma_f32_16x16x32_f16(false, a, false, b,
                                                (short)0, c, false, false);
}

static __device__ __forceinline__ float fast_exp2(float x) {
#if __has_builtin(__builtin_amdgcn_exp2f)
  return __builtin_amdgcn_exp2f(x);     // v_exp_f32, no range-check branches
#else
  return exp2f(x);
#endif
}
static __device__ __forceinline__ float fast_rcp(float x) {
#if __has_builtin(__builtin_amdgcn_rcpf)
  return __builtin_amdgcn_rcpf(x);      // v_rcp_f32
#else
  return 1.0f / x;
#endif
}
// sigmoid(x + b) with pre-scaled bias bs = -log2e*b; inf/0 saturation of exp2
// keeps it branch-free
static __device__ __forceinline__ float sigm_b(float x, float bs) {
  return fast_rcp(1.0f + fast_exp2(__builtin_fmaf(x, -LOG2E, bs)));
}
// tanh(x + b) with pre-scaled bias bs = 2*log2e*b
static __device__ __forceinline__ float tanh_b(float x, float bs) {
  return 1.0f - 2.0f * fast_rcp(1.0f + fast_exp2(__builtin_fmaf(x, LOG2E2, bs)));
}
// tanh(x), unbiased (cell output); native V_TANH_F32 if the builtin exists
static __device__ __forceinline__ float fast_tanh(float x) {
#if __has_builtin(__builtin_amdgcn_tanhf)
  return __builtin_amdgcn_tanhf(x);
#else
  return 1.0f - 2.0f * fast_rcp(1.0f + fast_exp2(x * LOG2E2));
#endif
}

static __device__ __forceinline__ v16h zero_v16h() {
  v16h v;
#pragma unroll
  for (int i = 0; i < 16; ++i) v[i] = (_Float16)0.0f;
  return v;
}
static __device__ __forceinline__ v8f zero_v8f() {
  v8f v;
#pragma unroll
  for (int i = 0; i < 8; ++i) v[i] = 0.0f;
  return v;
}
static __device__ __forceinline__ v16h cat8(v8h lo, v8h hi) {
  v16h v;
#pragma unroll
  for (int i = 0; i < 8; ++i) { v[i] = lo[i]; v[i + 8] = hi[i]; }
  return v;
}

// ---------------------------------------------------------------------------
// Weight prep: W (4H x K, row-major, fp32) -> f16 WMMA B-operand tiles.
// B-operand layout (16-bit, 32x16): lanes 0-15 hold K=0..7,16..23,
// lanes 16-31 hold K=8..15,24..31 (elem i -> K=8h+i, +8 if i>=8); N = lane%16.
// ws[((mat*8 + n)*32 + lane)*16 + i]; Wih0's K=13 padded with zeros to 32.
// ---------------------------------------------------------------------------
__global__ void prep_weights(const float* __restrict__ Wih0,
                             const float* __restrict__ Whh0,
                             const float* __restrict__ Wih1,
                             const float* __restrict__ Whh1,
                             _Float16* __restrict__ ws) {
  int idx = blockIdx.x * blockDim.x + threadIdx.x;  // 4*8*32*16 = 16384
  if (idx >= 4 * 8 * 32 * 16) return;
  int i    = idx & 15;
  int lane = (idx >> 4) & 31;
  int n    = (idx >> 9) & 7;
  int mat  = idx >> 12;
  int K    = ((lane < 16) ? 0 : 8) + i + ((i < 8) ? 0 : 8);
  int N    = lane & 15;
  int row  = n * 16 + N;
  const float* W;
  int Kdim;
  switch (mat) {
    case 0:  W = Wih0; Kdim = I_SZ; break;
    case 1:  W = Whh0; Kdim = H_SZ; break;
    case 2:  W = Wih1; Kdim = H_SZ; break;
    default: W = Whh1; Kdim = H_SZ; break;
  }
  float v = (K < Kdim) ? W[row * Kdim + K] : 0.0f;
  ws[idx] = (_Float16)v;
}

// ---------------------------------------------------------------------------
// Main fused kernel. Block = 128 threads = 4 waves = 2 batch tiles x 2 gate
// halves. Grid = 2048/32 = 64 blocks -> 256 concurrent recurrence chains.
// Gate layout in 4H: tiles 0-1 = i, 2-3 = f, 4-5 = g, 6-7 = o; a wave with
// gate-half gh owns tiles {gh, gh+2, gh+4, gh+6} == hidden columns gh*16+l16.
// ---------------------------------------------------------------------------
__global__ __launch_bounds__(128, 1) void lstm2_fused(
    const float* __restrict__ x,
    const float* __restrict__ bih0, const float* __restrict__ bhh0,
    const float* __restrict__ bih1, const float* __restrict__ bhh1,
    const float* __restrict__ Wc1,  const float* __restrict__ bc1,
    const float* __restrict__ Wc2,  const float* __restrict__ bc2,
    const _Float16* __restrict__ wB,
    float* __restrict__ out) {

  __shared__ __attribute__((aligned(32))) _Float16 sH[2][2][16 * 32]; // 4KB
  __shared__ __attribute__((aligned(16))) float    sHF[2][16 * 32];   // 4KB

  const int tid   = threadIdx.x;
  const int lane  = tid & 31;
  const int wave  = tid >> 5;
  const int btile = wave >> 1;                 // batch tile within block
  const int gh    = wave & 1;                  // gate-column half
  const int l16   = lane & 15;
  const int hi8   = (lane >> 4) << 3;          // row offset of this lane's elems
  const int KA0   = (lane < 16) ? 0 : 8;       // A-operand first K chunk
  const int b0    = blockIdx.x * 32 + btile * 16;
  const int col   = gh * 16 + l16;             // hidden column this lane owns

  // ---- pin all 16 B operands in VGPRs (loop-invariant, 128 VGPRs) ----
  v16h Bx0[4], Bh0[4], Bx1[4], Bh1[4];
#pragma unroll
  for (int q = 0; q < 4; ++q) {
    int n = gh + 2 * q;
    Bx0[q] = *(const v16h*)(wB + ((0 * 8 + n) * 32 + lane) * 16);
    Bh0[q] = *(const v16h*)(wB + ((1 * 8 + n) * 32 + lane) * 16);
    Bx1[q] = *(const v16h*)(wB + ((2 * 8 + n) * 32 + lane) * 16);
    Bh1[q] = *(const v16h*)(wB + ((3 * 8 + n) * 32 + lane) * 16);
  }

  // per-gate pre-scaled scalar biases (q: 0=i, 1=f, 2=g(tanh), 3=o)
  float bscl0[4], bscl1[4];
#pragma unroll
  for (int q = 0; q < 4; ++q) {
    int c = (gh + 2 * q) * 16 + l16;
    float b0v = bih0[c] + bhh0[c];
    float b1v = bih1[c] + bhh1[c];
    bscl0[q] = (q == 2) ? b0v * LOG2E2 : -b0v * LOG2E;
    bscl1[q] = (q == 2) ? b1v * LOG2E2 : -b1v * LOG2E;
  }

  v16h Ah0 = zero_v16h(), Ah1 = zero_v16h();
  float c0[8], c1[8], h1last[8];
#pragma unroll
  for (int r = 0; r < 8; ++r) { c0[r] = 0.f; c1[r] = 0.f; h1last[r] = 0.f; }

  const float* xp = x + (long long)(b0 + l16) * (T_SZ * I_SZ);

#pragma unroll 1
  for (int t = 0; t < T_SZ; ++t) {
    // ---- x_t as A-operand (K padded 13 -> 32 with zeros) ----
    v16h Ax = zero_v16h();
    {
      const float* xr = xp + t * I_SZ;
#pragma unroll
      for (int k = 0; k < 8; ++k) {
        int K = KA0 + k;
        if (K < I_SZ) Ax[k] = (_Float16)xr[K];
      }
      if (t + 16 < T_SZ) __builtin_prefetch(xr + 16 * I_SZ, 0, 3);
    }

    // ---- layer 0: 4 gate tiles, 2 WMMAs each, C starts at inline 0 ----
    v8f G[4];
#pragma unroll
    for (int q = 0; q < 4; ++q) {
      v8f acc = wmma_f16(Ax, Bx0[q], zero_v8f());
      G[q] = wmma_f16(Ah0, Bh0[q], acc);
    }
    float h0v[8];
#pragma unroll
    for (int r = 0; r < 8; ++r) {
      float iv = sigm_b(G[0][r], bscl0[0]);
      float fv = sigm_b(G[1][r], bscl0[1]);
      float gv = tanh_b(G[2][r], bscl0[2]);
      float ov = sigm_b(G[3][r], bscl0[3]);
      c0[r]  = fv * c0[r] + iv * gv;
      h0v[r] = ov * fast_tanh(c0[r]);
    }
    { // D-layout -> LDS row-major (f16)
      _Float16* hw = sH[btile][0];
#pragma unroll
      for (int r = 0; r < 8; ++r) hw[(hi8 + r) * 32 + col] = (_Float16)h0v[r];
    }
    __syncthreads();
    { // LDS -> A-operand layout (two 16B chunks per lane)
      const _Float16* hr = sH[btile][0] + l16 * 32;
      Ah0 = cat8(*(const v8h*)(hr + KA0), *(const v8h*)(hr + KA0 + 16));
    }

    // ---- layer 1: input = h0 (same A operand), recurrent = h1 ----
#pragma unroll
    for (int q = 0; q < 4; ++q) {
      v8f acc = wmma_f16(Ah0, Bx1[q], zero_v8f());
      G[q] = wmma_f16(Ah1, Bh1[q], acc);
    }
#pragma unroll
    for (int r = 0; r < 8; ++r) {
      float iv = sigm_b(G[0][r], bscl1[0]);
      float fv = sigm_b(G[1][r], bscl1[1]);
      float gv = tanh_b(G[2][r], bscl1[2]);
      float ov = sigm_b(G[3][r], bscl1[3]);
      c1[r]     = fv * c1[r] + iv * gv;
      h1last[r] = ov * fast_tanh(c1[r]);
    }
    {
      _Float16* hw = sH[btile][1];
#pragma unroll
      for (int r = 0; r < 8; ++r) hw[(hi8 + r) * 32 + col] = (_Float16)h1last[r];
    }
    __syncthreads();
    {
      const _Float16* hr = sH[btile][1] + l16 * 32;
      Ah1 = cat8(*(const v8h*)(hr + KA0), *(const v8h*)(hr + KA0 + 16));
    }
  }

  // ---- classifier head on final h1 (fp32 path, runs once; trivial cost) ----
#pragma unroll
  for (int r = 0; r < 8; ++r) sHF[btile][(hi8 + r) * 32 + col] = h1last[r];
  __syncthreads();

  if (gh == 0 && lane < 16) {
    const float* hv = sHF[btile] + l16 * 32;
    float hid[32];
#pragma unroll 4
    for (int j = 0; j < 32; ++j) {
      float s = bc1[j];
      for (int k = 0; k < 32; ++k) s += hv[k] * Wc1[j * 32 + k];
      hid[j] = fmaxf(s, 0.0f);
    }
#pragma unroll
    for (int c = 0; c < C_SZ; ++c) {
      float s = bc2[c];
      for (int j = 0; j < 32; ++j) s += hid[j] * Wc2[c * 32 + j];
      out[(b0 + l16) * C_SZ + c] = s;
    }
  }
}

extern "C" void kernel_launch(void* const* d_in, const int* in_sizes, int n_in,
                              void* d_out, int out_size, void* d_ws, size_t ws_size,
                              hipStream_t stream) {
  (void)in_sizes; (void)n_in; (void)out_size; (void)ws_size;
  const float* x    = (const float*)d_in[0];
  const float* Wih0 = (const float*)d_in[1];
  const float* Whh0 = (const float*)d_in[2];
  const float* bih0 = (const float*)d_in[3];
  const float* bhh0 = (const float*)d_in[4];
  const float* Wih1 = (const float*)d_in[5];
  const float* Whh1 = (const float*)d_in[6];
  const float* bih1 = (const float*)d_in[7];
  const float* bhh1 = (const float*)d_in[8];
  const float* Wc1  = (const float*)d_in[9];
  const float* bc1  = (const float*)d_in[10];
  const float* Wc2  = (const float*)d_in[11];
  const float* bc2  = (const float*)d_in[12];
  _Float16* ws = (_Float16*)d_ws;   // needs 32 KB of scratch
  float* out   = (float*)d_out;

  prep_weights<<<dim3(64), dim3(256), 0, stream>>>(Wih0, Whh0, Wih1, Whh1, ws);
  lstm2_fused<<<dim3(B_SZ / 32), dim3(128), 0, stream>>>(
      x, bih0, bhh0, bih1, bhh1, Wc1, bc1, Wc2, bc2, ws, out);
}
